// MultiHead_attn_91207925498490
// MI455X (gfx1250) — compile-verified, matrix operand order checked by reading, not run on hardware
//
#include <hip/hip_runtime.h>

#define DIMN 1024
#define HEADS 16
#define DH 64
#define BLK_LEN 128
#define BLK_NUM 32
#define BATCH 4
#define SEQ (BLK_NUM * BLK_LEN)          // 4096
#define ROWS (BATCH * SEQ)               // 16384

typedef __attribute__((ext_vector_type(8)))  float  v8f;
typedef __attribute__((ext_vector_type(8)))  __bf16 v8bf;
typedef __attribute__((ext_vector_type(16))) __bf16 v16bf;

// ---- workspace layout (bytes) ----
static const size_t WT_OFF  = 0;                         // W^T bf16, 3 * 2MB
static const size_t XBF_OFF = 6291456;                   // X bf16, 32MB
static const size_t Q_OFF   = XBF_OFF + 33554432ull;
static const size_t K_OFF   = Q_OFF   + 33554432ull;
static const size_t V_OFF   = K_OFF   + 33554432ull;
static const size_t O_OFF   = V_OFF   + 33554432ull;     // attn out f32, 64MB

__device__ __forceinline__ unsigned short f2bf(float f) {
    union { float f; unsigned u; } v; v.f = f;
    unsigned r = v.u + 0x7FFFu + ((v.u >> 16) & 1u);     // round-to-nearest-even
    return (unsigned short)(r >> 16);
}

__device__ __forceinline__ v16bf load_frag(const unsigned short* lo,
                                           const unsigned short* hi) {
    v8bf a = *(const v8bf*)lo;
    v8bf b = *(const v8bf*)hi;
    return __builtin_shufflevector(a, b, 0,1,2,3,4,5,6,7,8,9,10,11,12,13,14,15);
}

// CDNA5 async global->LDS copy (ASYNCcnt). LDS byte offset = low 32 bits of
// the generic shared-memory address (ISA 10.2 aperture rules).
__device__ __forceinline__ unsigned lds_addr(const void* p) {
    return (unsigned)(unsigned long long)p;
}
__device__ __forceinline__ void async_copy_b128(unsigned lds_off, const void* gptr) {
    asm volatile("global_load_async_to_lds_b128 %0, %1, off"
                 :: "v"(lds_off), "v"(gptr) : "memory");
}
__device__ __forceinline__ void wait_async0() {
    asm volatile("s_wait_asynccnt 0" ::: "memory");
}

// ---------------------------------------------------------------------------
// Kernel 0a: W fp32 -> bf16 with transpose: Wt[n][k] = bf16(W[k][n]).
// grid (32 ktiles, 32 ntiles, 3), 256 threads, 32x32 LDS tile.
// ---------------------------------------------------------------------------
__global__ __launch_bounds__(256)
void wconv_kernel(const float* __restrict__ w1,
                  const float* __restrict__ w2,
                  const float* __restrict__ w3,
                  unsigned short* __restrict__ wt) {
    const int z = blockIdx.z;
    const float* W = (z == 0) ? w1 : (z == 1) ? w2 : w3;
    unsigned short* Wt = wt + (size_t)z * DIMN * DIMN;

    const int k0 = blockIdx.x * 32;
    const int n0 = blockIdx.y * 32;
    __shared__ unsigned short t[32][33];

    const int tid = threadIdx.x;
    #pragma unroll
    for (int it = 0; it < 4; ++it) {
        int kk = it * 8 + (tid >> 5);
        int nn = tid & 31;
        t[kk][nn] = f2bf(W[(size_t)(k0 + kk) * DIMN + n0 + nn]);
    }
    __syncthreads();
    #pragma unroll
    for (int it = 0; it < 4; ++it) {
        int nn = it * 8 + (tid >> 5);
        int kk = tid & 31;
        Wt[(size_t)(n0 + nn) * DIMN + k0 + kk] = t[kk][nn];
    }
}

// ---------------------------------------------------------------------------
// Kernel 0b: X fp32 -> bf16 (so the GEMM can stage A with async byte copies)
// ---------------------------------------------------------------------------
__global__ __launch_bounds__(256)
void xconv_kernel(const float* __restrict__ X, unsigned short* __restrict__ Xbf) {
    size_t i = ((size_t)blockIdx.x * 256 + threadIdx.x) * 4;
    float4 xv = *(const float4*)(X + i);
    union { unsigned short s[4]; uint2 u; } p;
    p.s[0] = f2bf(xv.x); p.s[1] = f2bf(xv.y);
    p.s[2] = f2bf(xv.z); p.s[3] = f2bf(xv.w);
    *(uint2*)(Xbf + i) = p.u;
}

// ---------------------------------------------------------------------------
// Kernel 1: QKV projection GEMM. Xbf[16384,1024] * Wt^T + b -> bf16.
// Double-buffered async-to-LDS staging; 8 waves, 2x4 WMMA subtiles each.
// grid (8 Ntiles, 128 Mtiles, 3 matrices), 256 threads.
// ---------------------------------------------------------------------------
__global__ __launch_bounds__(256)
void proj_kernel(const unsigned short* __restrict__ Xbf,
                 const unsigned short* __restrict__ Wt,
                 const float* __restrict__ b1,
                 const float* __restrict__ b2,
                 const float* __restrict__ b3,
                 unsigned short* __restrict__ Q,
                 unsigned short* __restrict__ K,
                 unsigned short* __restrict__ V) {
    const int z = blockIdx.z;
    const unsigned short* Wz = Wt + (size_t)z * DIMN * DIMN;
    const float* bias = (z == 0) ? b1 : (z == 1) ? b2 : b3;
    unsigned short* Out = (z == 0) ? Q : (z == 1) ? K : V;

    const int n0 = blockIdx.x * 128;
    const int m0 = blockIdx.y * 128;

    __shared__ unsigned short As[2][128][32];   // 2 x 8KB
    __shared__ unsigned short Bs[2][128][32];   // 2 x 8KB (already [n][k])

    const int tid   = threadIdx.x;
    const int wave  = tid >> 5;
    const int lane  = tid & 31;
    const int wm    = wave >> 1;                // 0..3 (M)
    const int wn    = wave & 1;                 // 0..1 (N)
    const int lhalf = lane >> 4;
    const int l16   = lane & 15;

    // per-thread staging coords: two 16B chunks per tile
    const int sr = tid >> 2;                    // 0..63
    const int sc = (tid & 3) * 8;               // 0,8,16,24

    v8f acc[2][4] = {};

    // prologue fill of buffer 0
    {
        const int k0 = 0;
        #pragma unroll
        for (int it = 0; it < 2; ++it) {
            int r = it * 64 + sr;
            async_copy_b128(lds_addr(&As[0][r][sc]),
                            Xbf + (size_t)(m0 + r) * DIMN + k0 + sc);
            async_copy_b128(lds_addr(&Bs[0][r][sc]),
                            Wz + (size_t)(n0 + r) * DIMN + k0 + sc);
        }
    }
    wait_async0();
    __syncthreads();

    int buf = 0;
    for (int k0 = 0; k0 < DIMN; k0 += 32) {
        // async prefetch of next k-tile into the other buffer
        if (k0 + 32 < DIMN) {
            int nb = buf ^ 1;
            #pragma unroll
            for (int it = 0; it < 2; ++it) {
                int r = it * 64 + sr;
                async_copy_b128(lds_addr(&As[nb][r][sc]),
                                Xbf + (size_t)(m0 + r) * DIMN + (k0 + 32) + sc);
                async_copy_b128(lds_addr(&Bs[nb][r][sc]),
                                Wz + (size_t)(n0 + r) * DIMN + (k0 + 32) + sc);
            }
        }
        // compute on current buffer
        #pragma unroll
        for (int ms = 0; ms < 2; ++ms) {
            int ar = wm * 32 + ms * 16 + l16;
            v16bf a = load_frag(&As[buf][ar][lhalf * 8], &As[buf][ar][lhalf * 8 + 16]);
            #pragma unroll
            for (int ns = 0; ns < 4; ++ns) {
                int br = wn * 64 + ns * 16 + l16;
                v16bf b = load_frag(&Bs[buf][br][lhalf * 8], &Bs[buf][br][lhalf * 8 + 16]);
                acc[ms][ns] = __builtin_amdgcn_wmma_f32_16x16x32_bf16(
                    false, a, false, b, (short)0, acc[ms][ns], false, false);
            }
        }
        wait_async0();
        __syncthreads();
        buf ^= 1;
    }

    // epilogue: + bias, -> bf16
    #pragma unroll
    for (int ms = 0; ms < 2; ++ms) {
        #pragma unroll
        for (int ns = 0; ns < 4; ++ns) {
            #pragma unroll
            for (int g = 0; g < 8; ++g) {
                int row = m0 + wm * 32 + ms * 16 + lhalf * 8 + g;
                int col = n0 + wn * 64 + ns * 16 + l16;
                Out[(size_t)row * DIMN + col] = f2bf(acc[ms][ns][g] + bias[col]);
            }
        }
    }
}

// ---------------------------------------------------------------------------
// Kernel 2: block-local attention. grid (HEADS, BLK_NUM, BATCH), 256 threads.
// ---------------------------------------------------------------------------
__global__ __launch_bounds__(256)
void attn_kernel(const unsigned short* __restrict__ Qg,
                 const unsigned short* __restrict__ Kg,
                 const unsigned short* __restrict__ Vg,
                 const int* __restrict__ mask,
                 float* __restrict__ O) {
    const int head = blockIdx.x;
    const int blk  = blockIdx.y;
    const int b    = blockIdx.z;

    __shared__ unsigned short Qs[128][64];    // 16 KB
    __shared__ unsigned short Ks[128][64];    // 16 KB
    __shared__ unsigned short Vst[64][128];   // 16 KB, transposed [d][key]
    __shared__ unsigned short Ps[128][128];   // 32 KB probabilities bf16

    const int tid   = threadIdx.x;
    const int wave  = tid >> 5;               // query rows [wave*16, +16)
    const int lane  = tid & 31;
    const int lhalf = lane >> 4;
    const int l16   = lane & 15;

    const size_t rowbase = (size_t)b * SEQ + (size_t)blk * BLK_LEN;
    const int cb = head * DH;

    // stage Q, K via CDNA5 async-to-LDS; V through VGPRs (needs transpose)
    #pragma unroll
    for (int it = 0; it < 4; ++it) {
        int r = it * 32 + (tid >> 3);
        int d = (tid & 7) * 8;
        async_copy_b128(lds_addr(&Qs[r][d]), Qg + (rowbase + r) * DIMN + cb + d);
        async_copy_b128(lds_addr(&Ks[r][d]), Kg + (rowbase + r) * DIMN + cb + d);
        uint4 vv = *(const uint4*)(Vg + (rowbase + r) * DIMN + cb + d);
        const unsigned short* vs = (const unsigned short*)&vv;
        #pragma unroll
        for (int j = 0; j < 8; ++j) Vst[d + j][r] = vs[j];
    }
    wait_async0();
    __syncthreads();

    // S = Q K^T : 16 query rows x 128 keys per wave, K-dim 64 (2 steps)
    v8f s[8] = {};
    #pragma unroll
    for (int ks = 0; ks < 2; ++ks) {
        int ar = wave * 16 + l16;
        v16bf a = load_frag(&Qs[ar][ks * 32 + lhalf * 8],
                            &Qs[ar][ks * 32 + lhalf * 8 + 16]);
        #pragma unroll
        for (int nt = 0; nt < 8; ++nt) {
            int br = nt * 16 + l16;
            v16bf bb = load_frag(&Ks[br][ks * 32 + lhalf * 8],
                                 &Ks[br][ks * 32 + lhalf * 8 + 16]);
            s[nt] = __builtin_amdgcn_wmma_f32_16x16x32_bf16(
                false, a, false, bb, (short)0, s[nt], false, false);
        }
    }

    // scale + mask + row softmax (C-layout: vgpr g -> row g / 8+g per half)
    const float scale = 0.125f; // 1/sqrt(64)
    #pragma unroll
    for (int g = 0; g < 8; ++g) {
        int rloc = wave * 16 + lhalf * 8 + g;
        float mval = (float)mask[(rowbase + rloc) * DIMN + cb];
        float madd = (1.0f - mval) * -1.0e10f;
        float rmax = -3.0e38f;
        #pragma unroll
        for (int nt = 0; nt < 8; ++nt) {
            float x = s[nt][g] * scale + madd;
            s[nt][g] = x;
            rmax = fmaxf(rmax, x);
        }
        #pragma unroll
        for (int off = 1; off < 16; off <<= 1)
            rmax = fmaxf(rmax, __shfl_xor(rmax, off, 32));
        float rsum = 0.0f;
        #pragma unroll
        for (int nt = 0; nt < 8; ++nt) {
            float e = __expf(s[nt][g] - rmax);
            s[nt][g] = e;
            rsum += e;
        }
        #pragma unroll
        for (int off = 1; off < 16; off <<= 1)
            rsum += __shfl_xor(rsum, off, 32);
        float inv = 1.0f / rsum;
        #pragma unroll
        for (int nt = 0; nt < 8; ++nt)
            Ps[rloc][nt * 16 + l16] = f2bf(s[nt][g] * inv);
    }
    __syncthreads();

    // O = P V : 16 rows x 64 cols, K-dim 128 (4 steps)
    v8f o[4] = {};
    #pragma unroll
    for (int ks = 0; ks < 4; ++ks) {
        int ar = wave * 16 + l16;
        v16bf a = load_frag(&Ps[ar][ks * 32 + lhalf * 8],
                            &Ps[ar][ks * 32 + lhalf * 8 + 16]);
        #pragma unroll
        for (int nt = 0; nt < 4; ++nt) {
            int br = nt * 16 + l16; // d index (Vst transposed -> contiguous k)
            v16bf bb = load_frag(&Vst[br][ks * 32 + lhalf * 8],
                                 &Vst[br][ks * 32 + lhalf * 8 + 16]);
            o[nt] = __builtin_amdgcn_wmma_f32_16x16x32_bf16(
                false, a, false, bb, (short)0, o[nt], false, false);
        }
    }
    #pragma unroll
    for (int nt = 0; nt < 4; ++nt) {
        #pragma unroll
        for (int g = 0; g < 8; ++g) {
            int rloc = wave * 16 + lhalf * 8 + g;
            int col  = cb + nt * 16 + l16;
            O[(rowbase + rloc) * DIMN + col] = o[nt][g];
        }
    }
}

// ---------------------------------------------------------------------------
// Kernel 3: residual + LayerNorm. one block per row (16384 rows), 256 threads.
// ---------------------------------------------------------------------------
__global__ __launch_bounds__(256)
void ln_kernel(const float* __restrict__ O,
               const float* __restrict__ X,
               const float* __restrict__ gamma,
               const float* __restrict__ beta,
               float* __restrict__ out) {
    const size_t row = blockIdx.x;
    const int tid = threadIdx.x;
    const int c = tid * 4;

    float4 ov = *(const float4*)(O + row * DIMN + c);
    float4 xv = *(const float4*)(X + row * DIMN + c);
    float r0 = ov.x + xv.x, r1 = ov.y + xv.y, r2 = ov.z + xv.z, r3 = ov.w + xv.w;

    float ssum = r0 + r1 + r2 + r3;
    float ssq  = r0 * r0 + r1 * r1 + r2 * r2 + r3 * r3;
    #pragma unroll
    for (int off = 1; off < 32; off <<= 1) {
        ssum += __shfl_xor(ssum, off, 32);
        ssq  += __shfl_xor(ssq,  off, 32);
    }
    __shared__ float wsum[8], wsq[8];
    if ((tid & 31) == 0) { wsum[tid >> 5] = ssum; wsq[tid >> 5] = ssq; }
    __syncthreads();
    float tot = 0.0f, totq = 0.0f;
    #pragma unroll
    for (int i = 0; i < 8; ++i) { tot += wsum[i]; totq += wsq[i]; }

    const float mean = tot * (1.0f / DIMN);
    const float var  = totq * (1.0f / DIMN) - mean * mean;
    const float inv  = rsqrtf(var + 1.0e-3f);

    float4 res;
    res.x = (r0 - mean) * inv * gamma[c + 0] + beta[c + 0];
    res.y = (r1 - mean) * inv * gamma[c + 1] + beta[c + 1];
    res.z = (r2 - mean) * inv * gamma[c + 2] + beta[c + 2];
    res.w = (r3 - mean) * inv * gamma[c + 3] + beta[c + 3];
    *(float4*)(out + row * DIMN + c) = res;
}

// ---------------------------------------------------------------------------
extern "C" void kernel_launch(void* const* d_in, const int* in_sizes, int n_in,
                              void* d_out, int out_size, void* d_ws, size_t ws_size,
                              hipStream_t stream) {
    (void)in_sizes; (void)n_in; (void)out_size; (void)ws_size;

    const float* X     = (const float*)d_in[0];
    const int*   mask  = (const int*)d_in[1];
    const float* w1    = (const float*)d_in[2];
    const float* b1    = (const float*)d_in[3];
    const float* w2    = (const float*)d_in[4];
    const float* b2    = (const float*)d_in[5];
    const float* w3    = (const float*)d_in[6];
    const float* b3    = (const float*)d_in[7];
    const float* gamma = (const float*)d_in[8];
    const float* beta  = (const float*)d_in[9];
    float* out = (float*)d_out;

    char* ws = (char*)d_ws;
    unsigned short* Wt  = (unsigned short*)(ws + WT_OFF);
    unsigned short* Xbf = (unsigned short*)(ws + XBF_OFF);
    unsigned short* Q   = (unsigned short*)(ws + Q_OFF);
    unsigned short* K   = (unsigned short*)(ws + K_OFF);
    unsigned short* V   = (unsigned short*)(ws + V_OFF);
    float*          O   = (float*)(ws + O_OFF);

    // 0a) weights -> bf16, transposed to [n][k]
    wconv_kernel<<<dim3(DIMN / 32, DIMN / 32, 3), dim3(256), 0, stream>>>(w1, w2, w3, Wt);
    // 0b) activations -> bf16
    xconv_kernel<<<dim3((ROWS * DIMN) / 1024), dim3(256), 0, stream>>>(X, Xbf);

    // 1) Q/K/V projections (WMMA bf16, async-to-LDS double buffered)
    proj_kernel<<<dim3(DIMN / 128, ROWS / 128, 3), dim3(256), 0, stream>>>(
        Xbf, Wt, b1, b2, b3, Q, K, V);

    // 2) block-local attention (WMMA)
    attn_kernel<<<dim3(HEADS, BLK_NUM, BATCH), dim3(256), 0, stream>>>(
        Q, K, V, mask, O);

    // 3) residual + LayerNorm
    ln_kernel<<<dim3(ROWS), dim3(256), 0, stream>>>(O, X, gamma, beta, out);
}
